// MHSA_78168404787306
// MI455X (gfx1250) — matrix operand order repaired and checked by reading, not compiled, not run on hardware
//
#include <hip/hip_runtime.h>
#include <hip/hip_bf16.h>

// ---------------- problem constants (match reference) ----------------
#define Bc  2
#define Tc  2048
#define Hc  1024
#define NHc 16
#define Dc  64
#define Mc  (Bc * Tc)          // 4096 rows for the projection GEMMs
#define TEMPc 0.125f           // D^-0.5
#define NEGc  (-3.402823466e38f)

typedef _Float16 v16h __attribute__((ext_vector_type(16)));
typedef _Float16 h8   __attribute__((ext_vector_type(8)));
typedef float    v8f  __attribute__((ext_vector_type(8)));
typedef float    f4   __attribute__((ext_vector_type(4)));

union AF { v16h v; h8 h[2]; _Float16 e[16]; };

// ---------------- CDNA5 async global->LDS path (guarded) ----------------
#if defined(__has_builtin)
#if __has_builtin(__builtin_amdgcn_global_load_async_to_lds_b128)
#define HAVE_ASYNC_LDS 1
#endif
#endif

// builtin signature (from hipcc diagnostic): param0 = AS1 pointer to
// 16-byte int vector, param1 = LDS (AS3) pointer, then i32 offset, i32 cpol.
typedef int gv4i __attribute__((vector_size(4 * sizeof(int))));
typedef __attribute__((address_space(1))) gv4i* gv4i_gp;
typedef __attribute__((address_space(3))) gv4i* gv4i_lp;

__device__ __forceinline__ void cp16_async(const _Float16* g, _Float16* l) {
#if defined(HAVE_ASYNC_LDS)
  __builtin_amdgcn_global_load_async_to_lds_b128(
      (gv4i_gp)(size_t)(const void*)g,          // inttoptr -> AS1
      (gv4i_lp)(unsigned)(size_t)(void*)l,      // low 32 bits = LDS offset
      0, 0);
#else
  *(h8*)l = *(const h8*)g;                      // fallback: load + ds_store
#endif
}

__device__ __forceinline__ void wait_async() {
#if defined(HAVE_ASYNC_LDS)
#if __has_builtin(__builtin_amdgcn_s_wait_asynccnt)
  __builtin_amdgcn_s_wait_asynccnt(0);
#else
  asm volatile("s_wait_asynccnt 0x0" ::: "memory");
#endif
#endif
}

// D = A(16x32 f16) * B(32x16 f16) + C(16x16 f32)
__device__ __forceinline__ v8f wmma16(v16h a, v16h b, v8f c) {
  return __builtin_amdgcn_wmma_f32_16x16x32_f16(
      /*neg_a=*/false, a, /*neg_b=*/false, b,
      /*c_mod=*/(short)0, c, /*reuse_a=*/false, /*reuse_b=*/false);
}

// ---- A-matrix fragment (16x32 MxK f16): lane L holds row M=(L&15);
// lanes<16: K=kk..kk+15; lanes>=16: K=kk+16..kk+31. Works for
// global (global_load_b128) and LDS (ds_load_b128) source pointers.
__device__ __forceinline__ v16h load_a_f16(const _Float16* base, int ld,
                                           int row, int kk, int lane) {
  const int sel = lane >> 4;
  const _Float16* p = base + (size_t)(row + (lane & 15)) * ld + kk + sel * 16;
  AF u;
  u.h[0] = *(const h8*)(p);
  u.h[1] = *(const h8*)(p + 8);
  return u.v;
}

// ---- B-matrix fragment (32x16 KxN f16): lane L holds column N=(L&15);
// lanes<16: K=kk..kk+15; lanes>=16: K=kk+16..kk+31 (16 contiguous halves).
__device__ __forceinline__ v16h load_b_f16(const _Float16* base, int ld,
                                           int col, int kk, int lane) {
  const int sel = lane >> 4;
  const _Float16* p = base + (size_t)(col + (lane & 15)) * ld + kk + sel * 16;
  AF u;
  u.h[0] = *(const h8*)(p);
  u.h[1] = *(const h8*)(p + 8);
  return u.v;
}

// =====================================================================
// fp32 -> f16 conversion (done ONCE per tensor; GEMMs then run pure f16)
// =====================================================================
__global__ __launch_bounds__(256) void cvt_f16(const float* __restrict__ x,
                                               _Float16* __restrict__ y) {
  const size_t i = ((size_t)blockIdx.x * blockDim.x + threadIdx.x) * 8;
  f4 a = *(const f4*)(x + i);
  f4 b = *(const f4*)(x + i + 4);
  h8 o;
#pragma unroll
  for (int j = 0; j < 4; ++j) { o[j] = (_Float16)a[j]; o[4 + j] = (_Float16)b[j]; }
  *(h8*)(y + i) = o;
}

// =====================================================================
// GEMM: Y[M,N] = X[M,K] @ W[N,K]^T (+ bias). One wave owns a 32x64 tile:
// 2 A-frags x 4 B-frags -> 8 WMMAs per 32-wide K step.
// OMODE: 0 = f16 row-major, 1 = f32 + bias (final output),
//        2 = f16 scattered to per-head V^T layout (B,NH,D,T).
// =====================================================================
template <int OMODE>
__global__ __launch_bounds__(128) void gemm_xwT(const _Float16* __restrict__ A,
                                                const _Float16* __restrict__ W,
                                                const float* __restrict__ bias,
                                                void* __restrict__ Cout,
                                                int Ndim, int Kdim) {
  const int lane = threadIdx.x & 31;
  const int wave = blockIdx.x * (blockDim.x >> 5) + (threadIdx.x >> 5);
  const int ntiles = Ndim >> 6;
  const int tm = (wave / ntiles) * 32;
  const int tn = (wave % ntiles) * 64;

  v8f acc[2][4] = {};
  for (int kk = 0; kk < Kdim; kk += 32) {
    const v16h a0 = load_a_f16(A, Kdim, tm,      kk, lane);
    const v16h a1 = load_a_f16(A, Kdim, tm + 16, kk, lane);
#pragma unroll
    for (int nt = 0; nt < 4; ++nt) {
      const v16h b = load_b_f16(W, Kdim, tn + nt * 16, kk, lane);
      acc[0][nt] = wmma16(a0, b, acc[0][nt]);
      acc[1][nt] = wmma16(a1, b, acc[1][nt]);
    }
  }

  // C/D layout: lane holds column n=(lane&15); rows m = 8*(lane>>4)+r.
  const int sel = lane >> 4;
  const int cn  = lane & 15;
#pragma unroll
  for (int hf = 0; hf < 2; ++hf) {
#pragma unroll
    for (int nt = 0; nt < 4; ++nt) {
      const int col = tn + nt * 16 + cn;
      const float bv = (OMODE == 1) ? bias[col] : 0.0f;
#pragma unroll
      for (int r = 0; r < 8; ++r) {
        const int row = tm + hf * 16 + sel * 8 + r;   // token index
        const float v = acc[hf][nt][r] + bv;
        if (OMODE == 0) {
          ((_Float16*)Cout)[(size_t)row * Ndim + col] = (_Float16)v;
        } else if (OMODE == 1) {
          ((float*)Cout)[(size_t)row * Ndim + col] = v;
        } else {            // V^T per head: Vt[b,h,chan,t]
          const int b = row >> 11, t = row & (Tc - 1);
          const int h = col >> 6,  ch = col & (Dc - 1);
          ((_Float16*)Cout)[(((size_t)b * NHc + h) * Dc + ch) * Tc + t] =
              (_Float16)v;
        }
      }
    }
  }
}

// =====================================================================
// Flash attention. Block = 4 waves sharing one (b,h); wave w owns queries
// [qb64 + 16w, +16). K/V tiles (32 keys) are staged cooperatively into LDS
// with async global->LDS copies, double buffered:
//   issue tile j+1 -> compute tile j from LDS -> s_wait_asynccnt 0 -> barrier.
// S^T = K * Q^T, online softmax per query column (one shfl_xor(16)),
// O^T = V^T * P^T accumulated in registers.
// =====================================================================
__global__ __launch_bounds__(128) void attn(const _Float16* __restrict__ Qh,
                                            const _Float16* __restrict__ Kh,
                                            const _Float16* __restrict__ Vt,
                                            const unsigned char* __restrict__ mask,
                                            _Float16* __restrict__ Oh) {
  __shared__ _Float16 ldsK[2][32 * 64];   // [key][chan]
  __shared__ _Float16 ldsV[2][64 * 32];   // [chan][key]

  const int lane = threadIdx.x & 31;
  const int warp = threadIdx.x >> 5;
  const int tid  = threadIdx.x;

  const int q64 = blockIdx.x % (Tc / 64);
  const int bh  = blockIdx.x / (Tc / 64);
  const int h = bh % NHc;
  const int b = bh / NHc;
  const int qbase = q64 * 64 + warp * 16;
  const int sel = lane >> 4;

  const _Float16* Qp = Qh + (size_t)b * Tc * Hc + h * Dc;    // row t: +t*Hc
  const _Float16* Kp = Kh + (size_t)b * Tc * Hc + h * Dc;
  const _Float16* Vp = Vt + ((size_t)b * NHc + h) * Dc * Tc; // row ch: +ch*Tc
  const int qg = qbase + (lane & 15);
  const unsigned char* mrow = mask + (size_t)b * Tc * Tc + (size_t)qg * Tc;

  // Q^T B-fragments, cached for the whole key loop.
  const v16h qb0 = load_b_f16(Qp, Hc, qbase, 0, lane);
  const v16h qb1 = load_b_f16(Qp, Hc, qbase, 32, lane);

  // cooperative staging: 256 16B chunks for K + 256 for V, 128 threads.
  auto stage = [&](int buf, int keyb) {
#pragma unroll
    for (int pass = 0; pass < 2; ++pass) {
      const int c = tid + pass * 128;
      cp16_async(Kp + (size_t)(keyb + (c >> 3)) * Hc + (c & 7) * 8,
                 &ldsK[buf][c * 8]);
      cp16_async(Vp + (size_t)(c >> 2) * Tc + keyb + (c & 3) * 8,
                 &ldsV[buf][c * 8]);
    }
  };

  v8f ot[4] = {};                        // O^T: 4 chan-tiles of 16x16
  float mrun = -3.0e38f;
  float lrun = 0.0f;

  stage(0, 0);
  wait_async();
  __syncthreads();

  const int nblk = Tc / 32;
  for (int j = 0; j < nblk; ++j) {
    const int buf = j & 1;
    if (j + 1 < nblk) stage(buf ^ 1, (j + 1) * 32);

    const _Float16* kt = ldsK[buf];
    const _Float16* vt = ldsV[buf];

    // S^T tiles (rows = keys, cols = queries) out of LDS.
    v16h ka;
    v8f s0 = {}, s1 = {};
    ka = load_a_f16(kt, 64, 0, 0, lane);   s0 = wmma16(ka, qb0, s0);
    ka = load_a_f16(kt, 64, 0, 32, lane);  s0 = wmma16(ka, qb1, s0);
    ka = load_a_f16(kt, 64, 16, 0, lane);  s1 = wmma16(ka, qb0, s1);
    ka = load_a_f16(kt, 64, 16, 32, lane); s1 = wmma16(ka, qb1, s1);

    // mask (f32-min fill BEFORE temperature scaling, as in reference);
    // 8 mask bytes per half-tile fetched as one u64.
    const int keyb = j * 32;
    const unsigned long long m0 =
        *(const unsigned long long*)(mrow + keyb + sel * 8);
    const unsigned long long m1 =
        *(const unsigned long long*)(mrow + keyb + 16 + sel * 8);
#pragma unroll
    for (int r = 0; r < 8; ++r) {
      const float x0 = ((m0 >> (8 * r)) & 0xffULL) ? s0[r] : NEGc;
      const float x1 = ((m1 >> (8 * r)) & 0xffULL) ? s1[r] : NEGc;
      s0[r] = x0 * TEMPc;
      s1[r] = x1 * TEMPc;
    }

    // online softmax over the 32 keys of this block (per query column)
    float tmax = -3.0e38f;
#pragma unroll
    for (int r = 0; r < 8; ++r) tmax = fmaxf(tmax, fmaxf(s0[r], s1[r]));
    tmax = fmaxf(tmax, __shfl_xor(tmax, 16, 32));
    const float mnew = fmaxf(mrun, tmax);
    const float alpha = __expf(mrun - mnew);

    float rsum = 0.0f;
#pragma unroll
    for (int r = 0; r < 8; ++r) {
      s0[r] = __expf(s0[r] - mnew);
      s1[r] = __expf(s1[r] - mnew);
      rsum += s0[r] + s1[r];
    }
    rsum += __shfl_xor(rsum, 16, 32);
    lrun = lrun * alpha + rsum;
    mrun = mnew;

#pragma unroll
    for (int ct = 0; ct < 4; ++ct)
#pragma unroll
      for (int r = 0; r < 8; ++r) ot[ct][r] *= alpha;

    // P^T B-fragment (32 keys x 16 queries): lane and lane^16 hold the same
    // query column with complementary key halves -> exchange via shfl_xor.
    AF pT;
#pragma unroll
    for (int r = 0; r < 8; ++r) {
      const float o0 = __shfl_xor(s0[r], 16, 32);
      const float o1 = __shfl_xor(s1[r], 16, 32);
      pT.e[r]     = (_Float16)((lane < 16) ? s0[r] : o1);
      pT.e[8 + r] = (_Float16)((lane < 16) ? o0    : s1[r]);
    }

    // O^T += V^T(chan-tile x 32 keys) * P^T   (A-frags out of LDS)
#pragma unroll
    for (int ct = 0; ct < 4; ++ct) {
      const v16h va = load_a_f16(vt, 32, ct * 16, 0, lane);
      ot[ct] = wmma16(va, pT.v, ot[ct]);
    }

    wait_async();
    __syncthreads();
  }

  // normalize and store O (back to [B,T,H] f16 for the output projection)
  const float inv = 1.0f / lrun;
  _Float16* orow = Oh + ((size_t)b * Tc + qg) * Hc + h * Dc;
#pragma unroll
  for (int ct = 0; ct < 4; ++ct)
#pragma unroll
    for (int r = 0; r < 8; ++r)
      orow[ct * 16 + sel * 8 + r] = (_Float16)(ot[ct][r] * inv);
}

// =====================================================================
// launch
// =====================================================================
extern "C" void kernel_launch(void* const* d_in, const int* in_sizes, int n_in,
                              void* d_out, int out_size, void* d_ws, size_t ws_size,
                              hipStream_t stream) {
  (void)in_sizes; (void)n_in; (void)out_size; (void)ws_size;

  const float* q    = (const float*)d_in[0];
  const float* k    = (const float*)d_in[1];
  const float* v    = (const float*)d_in[2];
  const unsigned char* mask = (const unsigned char*)d_in[3];
  const float* Wq   = (const float*)d_in[4];
  const float* Wk   = (const float*)d_in[5];
  const float* Wv   = (const float*)d_in[6];
  const float* Wo   = (const float*)d_in[7];
  const float* bo   = (const float*)d_in[8];
  float* out = (float*)d_out;

  // workspace layout (f16 tensors)
  char* ws = (char*)d_ws;
  const size_t szX = (size_t)Mc * Hc * sizeof(_Float16);   // 8 MB
  const size_t szW = (size_t)Hc * Hc * sizeof(_Float16);   // 2 MB
  _Float16* Xq  = (_Float16*)(ws);
  _Float16* Xk  = (_Float16*)(ws + szX);
  _Float16* Xv  = (_Float16*)(ws + 2 * szX);
  _Float16* Wqh = (_Float16*)(ws + 3 * szX);
  _Float16* Wkh = (_Float16*)(ws + 3 * szX + szW);
  _Float16* Wvh = (_Float16*)(ws + 3 * szX + 2 * szW);
  _Float16* Woh = (_Float16*)(ws + 3 * szX + 3 * szW);
  _Float16* Qh  = (_Float16*)(ws + 3 * szX + 4 * szW);
  _Float16* Kh  = (_Float16*)(ws + 4 * szX + 4 * szW);
  _Float16* Vt  = (_Float16*)(ws + 5 * szX + 4 * szW);     // (B,NH,D,T)
  _Float16* Oh  = (_Float16*)(ws + 6 * szX + 4 * szW);

  // one-time fp32 -> f16 conversions
  const int nX = Mc * Hc, nW = Hc * Hc;
  cvt_f16<<<nX / (256 * 8), 256, 0, stream>>>(q, Xq);
  cvt_f16<<<nX / (256 * 8), 256, 0, stream>>>(k, Xk);
  cvt_f16<<<nX / (256 * 8), 256, 0, stream>>>(v, Xv);
  cvt_f16<<<nW / (256 * 8), 256, 0, stream>>>(Wq, Wqh);
  cvt_f16<<<nW / (256 * 8), 256, 0, stream>>>(Wk, Wkh);
  cvt_f16<<<nW / (256 * 8), 256, 0, stream>>>(Wv, Wvh);
  cvt_f16<<<nW / (256 * 8), 256, 0, stream>>>(Wo, Woh);

  // projections: (Mc/32)*(Hc/64) = 2048 waves, 4 per block
  const dim3 gblk(128);
  const dim3 ggrd((Mc / 32) * (Hc / 64) / 4);
  gemm_xwT<0><<<ggrd, gblk, 0, stream>>>(Xq, Wqh, nullptr, Qh, Hc, Hc);
  gemm_xwT<0><<<ggrd, gblk, 0, stream>>>(Xk, Wkh, nullptr, Kh, Hc, Hc);
  gemm_xwT<2><<<ggrd, gblk, 0, stream>>>(Xv, Wvh, nullptr, Vt, Hc, Hc);  // -> V^T

  // flash attention: one block per (b,h,64 queries)
  attn<<<Bc * NHc * (Tc / 64), 128, 0, stream>>>(Qh, Kh, Vt, mask, Oh);

  // output projection with bias -> f32
  gemm_xwT<1><<<ggrd, gblk, 0, stream>>>(Oh, Woh, bo, out, Hc, Hc);
}
